// RSS_Attn_SS2D_Vmamba_6493990552154
// MI455X (gfx1250) — matrix operand order repaired and checked by reading, not compiled, not run on hardware
//
#include <hip/hip_runtime.h>
#include <hip/hip_bf16.h>
#include <math.h>

// ---------------- problem constants ----------------
#define B_DIM   4
#define C_DIM   96
#define H_DIM   64
#define W_DIM   64
#define L_TOT   4096           // H*W
#define D_DIM   192            // C*EXPAND
#define N_DIM   16
#define R_DIM   6
#define K_DIR   4
#define HID_DIM 115
#define CPROJ   38             // R + 2N

typedef __attribute__((ext_vector_type(16))) _Float16 v16h;
typedef __attribute__((ext_vector_type(8)))  float    v8f;

// WMMA f32 = f16 x f16 + f32  (16x16x32), probe-confirmed signature
__device__ __forceinline__ v8f wmma_f16(v16h a, v16h b, v8f c) {
  return __builtin_amdgcn_wmma_f32_16x16x32_f16(false, a, false, b, (short)0, c,
                                                false, false);
}

// A-fragment (16x32 f16, MxK): lane row = lane&15, element e covers
//   k = (e<8 ? e : e+8) + 8*(lane>>4)
__device__ __forceinline__ int ak_off(int e, int half) {
  return (e < 8 ? e : e + 8) + 8 * half;
}
// B-fragment (32x16 f16, KxN): lane col = lane&15, element e covers
//   k = (e<8 ? e : e+8) + 8*(lane>>4)
__device__ __forceinline__ int bk_off(int e, int half) {
  return (e < 8 ? e : e + 8) + 8 * half;
}

__device__ __forceinline__ float sigf(float v) {
  return 1.0f / (1.0f + __expf(-v));
}

// ---- CDNA5 async global->LDS copies (ASYNCcnt-tracked, ISA ch10/§15.18) ----
// VDST operand carries the LDS byte address; flat pointers to __shared__ carry
// the LDS offset in their low 32 bits (aperture layout, ISA 10.2).
__device__ __forceinline__ void async_ld_b128(void* lds, const void* g) {
  asm volatile("global_load_async_to_lds_b128 %0, %1, off"
               :: "v"((uint32_t)(uintptr_t)lds),
                  "v"((uint64_t)(uintptr_t)g) : "memory");
}
__device__ __forceinline__ void async_ld_b32(void* lds, const void* g) {
  asm volatile("global_load_async_to_lds_b32 %0, %1, off"
               :: "v"((uint32_t)(uintptr_t)lds),
                  "v"((uint64_t)(uintptr_t)g) : "memory");
}
__device__ __forceinline__ void wait_async0() {
  asm volatile("s_wait_asynccnt 0" ::: "memory");
}

// Map xs[b,k,d,l] -> spatial index into x1 (B,D,H,W); H==W==64 so the
// hw<->wh transpose is the same bit swap both ways.
__device__ __forceinline__ int scan_index(int k, int l) {
  int ll = (k >= 2) ? (L_TOT - 1 - l) : l;
  if (k & 1) ll = ((ll & (W_DIM - 1)) << 6) | (ll >> 6);
  return ll;
}

// ============ K1: in_proj GEMM [16384 x 96] x [96 x 384] ============
// A tile staged k-major in LDS via async B128; shared by the block's 4 waves.
__global__ void k_inproj(const float* __restrict__ x,
                         const float* __restrict__ w,
                         const float* __restrict__ bias,
                         float* __restrict__ x1pre,
                         float* __restrict__ z) {
  __shared__ float sA[C_DIM * 16];             // [k][row], 6 KB
  int lane = threadIdx.x & 31, wave = threadIdx.x >> 5;
  int tile_m = blockIdx.x;
  int m0 = tile_m * 16;
  int b = m0 >> 12;
  int lbase = m0 & (L_TOT - 1);

  // stage A: for each k, 16 contiguous floats (rows) -> 4 x B128
  for (int i = threadIdx.x; i < C_DIM * 4; i += 128) {
    int k = i >> 2, part = i & 3;
    async_ld_b128(&sA[k * 16 + part * 4],
                  &x[(b * C_DIM + k) * L_TOT + lbase + part * 4]);
  }
  wait_async0();
  __syncthreads();

  int tile_n = blockIdx.y * 4 + wave;          // 24 n-tiles total
  int half = lane >> 4, rc = lane & 15;
  int n = tile_n * 16 + rc;
  v8f acc = {};
  for (int kb = 0; kb < C_DIM; kb += 32) {
    v16h af, bf;
#pragma unroll
    for (int e = 0; e < 16; ++e)
      af[e] = (_Float16)sA[(kb + ak_off(e, half)) * 16 + rc];
#pragma unroll
    for (int e = 0; e < 16; ++e)
      bf[e] = (_Float16)w[n * C_DIM + kb + bk_off(e, half)];
    acc = wmma_f16(af, bf, acc);
  }
  float bv = bias[n];
#pragma unroll
  for (int r = 0; r < 8; ++r) {
    int m = m0 + r + 8 * half;
    int l = lbase + r + 8 * half;
    float v = acc[r] + bv;
    if (n < D_DIM) x1pre[(b * D_DIM + n) * L_TOT + l] = v;
    else           z[m * D_DIM + (n - D_DIM)] = v;
  }
}

// ============ K2: depthwise 3x3 conv + bias + SiLU ============
__global__ void k_conv(const float* __restrict__ xin,
                       const float* __restrict__ cw,
                       const float* __restrict__ cb,
                       float* __restrict__ xout) {
  int idx = blockIdx.x * blockDim.x + threadIdx.x;
  if (idx >= B_DIM * D_DIM * L_TOT) return;
  int l = idx & (L_TOT - 1);
  int bd = idx >> 12;
  int d = bd % D_DIM;
  int h = l >> 6, wcol = l & 63;
  float s = cb[d];
#pragma unroll
  for (int dy = -1; dy <= 1; ++dy) {
#pragma unroll
    for (int dx = -1; dx <= 1; ++dx) {
      int hh = h + dy, ww = wcol + dx;
      if (hh >= 0 && hh < H_DIM && ww >= 0 && ww < W_DIM)
        s += xin[(bd << 12) + (hh << 6) + ww] * cw[d * 9 + (dy + 1) * 3 + (dx + 1)];
    }
  }
  xout[idx] = s * sigf(s);
}

// ============ K3: x_proj GEMM per direction: [16384 x 192] x [192 x 38] ====
// A tile is a scan-order gather; staged per-lane with async B32.
__global__ void k_xproj(const float* __restrict__ x1,
                        const float* __restrict__ w,
                        float* __restrict__ xdbl) {
  __shared__ float sA[D_DIM * 16];             // [d][row], 12 KB
  int lane = threadIdx.x & 31, wave = threadIdx.x >> 5;
  int tile_m = blockIdx.x;
  int kdir = blockIdx.z;
  int m0 = tile_m * 16;
  int b = m0 >> 12;
  int lbase = m0 & (L_TOT - 1);

  for (int i = threadIdx.x; i < D_DIM * 16; i += 128) {
    int d = i >> 4, row = i & 15;
    int sidx = scan_index(kdir, lbase + row);
    async_ld_b32(&sA[i], &x1[(b * D_DIM + d) * L_TOT + sidx]);
  }
  wait_async0();
  __syncthreads();

  int tile_n = blockIdx.y * 4 + wave;
  if (tile_n >= 3) return;                     // 38 -> 3 tiles of 16
  int half = lane >> 4, rc = lane & 15;
  int c = tile_n * 16 + rc;
  v8f acc = {};
  for (int kb = 0; kb < D_DIM; kb += 32) {
    v16h af, bf;
#pragma unroll
    for (int e = 0; e < 16; ++e)
      af[e] = (_Float16)sA[(kb + ak_off(e, half)) * 16 + rc];
#pragma unroll
    for (int e = 0; e < 16; ++e) {
      int d = kb + bk_off(e, half);
      bf[e] = (_Float16)((c < CPROJ) ? w[(kdir * CPROJ + c) * D_DIM + d] : 0.0f);
    }
    acc = wmma_f16(af, bf, acc);
  }
  if (c < CPROJ) {
#pragma unroll
    for (int r = 0; r < 8; ++r) {
      int l = lbase + r + 8 * half;
      xdbl[((b * K_DIR + kdir) * CPROJ + c) * L_TOT + l] = acc[r];
    }
  }
}

// ============ K4: selective scan; dt-projection + softplus fused ============
__global__ void k_scan(const float* __restrict__ x1,
                       const float* __restrict__ xdbl,
                       const float* __restrict__ dtw,
                       const float* __restrict__ dtb,
                       const float* __restrict__ alog,
                       const float* __restrict__ ds,
                       float* __restrict__ ys) {
  int idx = blockIdx.x * blockDim.x + threadIdx.x;
  if (idx >= B_DIM * K_DIR * D_DIM) return;
  int d = idx % D_DIM;
  int bk = idx / D_DIM;
  int k = bk % K_DIR;
  int b = bk / K_DIR;
  int kd = k * D_DIM + d;

  float Av[N_DIM], wr[R_DIM];
#pragma unroll
  for (int n = 0; n < N_DIM; ++n) Av[n] = -__expf(alog[kd * N_DIM + n]);
#pragma unroll
  for (int r = 0; r < R_DIM; ++r) wr[r] = dtw[kd * R_DIM + r];
  float bias = dtb[kd], Dv = ds[kd];

  const float* xd = xdbl + (size_t)(b * K_DIR + k) * CPROJ * L_TOT;
  const float* ub = x1 + (size_t)(b * D_DIM + d) * L_TOT;
  float* yo = ys + (size_t)((b * K_DIR + k) * D_DIM + d) * L_TOT;

  float h[N_DIM];
#pragma unroll
  for (int n = 0; n < N_DIM; ++n) h[n] = 0.0f;

  for (int l = 0; l < L_TOT; ++l) {
    float t = bias;
#pragma unroll
    for (int r = 0; r < R_DIM; ++r) t += xd[r * L_TOT + l] * wr[r];
    float dt = (t > 20.0f) ? t : log1pf(__expf(t));   // softplus
    float u = ub[scan_index(k, l)];
    float du = dt * u;
    float y = 0.0f;
#pragma unroll
    for (int n = 0; n < N_DIM; ++n) {
      h[n] = h[n] * __expf(dt * Av[n]) + du * xd[(R_DIM + n) * L_TOT + l];
      y += h[n] * xd[(R_DIM + N_DIM + n) * L_TOT + l];
    }
    yo[l] = y + Dv * u;
  }
}

// ============ K5: combine 4 directions + LayerNorm + SiLU(z) gate ==========
__global__ void k_combine(const float* __restrict__ ys,
                          const float* __restrict__ z,
                          const float* __restrict__ g,
                          const float* __restrict__ be,
                          float* __restrict__ yout) {
  int lane = threadIdx.x & 31;
  int wave = threadIdx.x >> 5;
  int m = blockIdx.x * (blockDim.x >> 5) + wave;
  if (m >= B_DIM * L_TOT) return;
  int b = m >> 12, l = m & (L_TOT - 1);
  int l1 = ((l & 63) << 6) | (l >> 6);          // wh index of this hw pos
  float vv[6], s = 0.0f, q = 0.0f;
#pragma unroll
  for (int j = 0; j < 6; ++j) {
    int d = lane + 32 * j;
    float y0 = ys[((size_t)((b * 4 + 0) * D_DIM + d)) * L_TOT + l];
    float y1 = ys[((size_t)((b * 4 + 1) * D_DIM + d)) * L_TOT + l1];
    float y2 = ys[((size_t)((b * 4 + 2) * D_DIM + d)) * L_TOT + (L_TOT - 1 - l)];
    float y3 = ys[((size_t)((b * 4 + 3) * D_DIM + d)) * L_TOT + (L_TOT - 1 - l1)];
    float v = y0 + y1 + y2 + y3;
    vv[j] = v; s += v; q += v * v;
  }
#pragma unroll
  for (int off = 16; off > 0; off >>= 1) {
    s += __shfl_xor(s, off, 32);
    q += __shfl_xor(q, off, 32);
  }
  float mu = s * (1.0f / D_DIM);
  float var = q * (1.0f / D_DIM) - mu * mu;
  float rs = rsqrtf(var + 1e-5f);
#pragma unroll
  for (int j = 0; j < 6; ++j) {
    int d = lane + 32 * j;
    float zv = z[(size_t)m * D_DIM + d];
    float yn = (vv[j] - mu) * rs * g[d] + be[d];
    yout[(size_t)m * D_DIM + d] = yn * zv * sigf(zv);
  }
}

// ============ K6: out_proj GEMM [16384 x 192] x [192 x 96] + residual ======
// A tile (16x192, fully contiguous in Y) staged via async B128.
__global__ void k_outproj(const float* __restrict__ y,
                          const float* __restrict__ w,
                          const float* __restrict__ bias,
                          const float* __restrict__ x,
                          float* __restrict__ x2) {
  __shared__ float sA[16 * D_DIM];             // [row][k], 12 KB
  int lane = threadIdx.x & 31, wave = threadIdx.x >> 5;
  int tile_m = blockIdx.x;
  int m0 = tile_m * 16;
  int b = m0 >> 12;
  int lbase = m0 & (L_TOT - 1);

  for (int i = threadIdx.x; i < 16 * D_DIM / 4; i += 128)
    async_ld_b128(&sA[i * 4], &y[m0 * D_DIM + i * 4]);
  wait_async0();
  __syncthreads();

  int tile_n = blockIdx.y * 4 + wave;
  if (tile_n >= 6) return;
  int half = lane >> 4, rc = lane & 15;
  int n = tile_n * 16 + rc;
  v8f acc = {};
  for (int kb = 0; kb < D_DIM; kb += 32) {
    v16h af, bf;
#pragma unroll
    for (int e = 0; e < 16; ++e)
      af[e] = (_Float16)sA[rc * D_DIM + kb + ak_off(e, half)];
#pragma unroll
    for (int e = 0; e < 16; ++e)
      bf[e] = (_Float16)w[n * D_DIM + kb + bk_off(e, half)];
    acc = wmma_f16(af, bf, acc);
  }
  float bv = bias[n];
#pragma unroll
  for (int r = 0; r < 8; ++r) {
    int m = m0 + r + 8 * half;
    int l = lbase + r + 8 * half;
    x2[m * C_DIM + n] = acc[r] + bv + x[(b * C_DIM + n) * L_TOT + l];
  }
}

// ============ K7: mlp1 GEMM [16384 x 96] x [96 x 115] + BN1 + SiLU =========
__global__ void k_mlp1(const float* __restrict__ x2,
                       const float* __restrict__ w,
                       const float* __restrict__ g,
                       const float* __restrict__ be,
                       const float* __restrict__ mm,
                       const float* __restrict__ vv,
                       float* __restrict__ h1) {
  __shared__ float sA[16 * C_DIM];             // [row][k], 6 KB
  int lane = threadIdx.x & 31, wave = threadIdx.x >> 5;
  int tile_m = blockIdx.x;
  int m0 = tile_m * 16;

  for (int i = threadIdx.x; i < 16 * C_DIM / 4; i += 128)
    async_ld_b128(&sA[i * 4], &x2[m0 * C_DIM + i * 4]);
  wait_async0();
  __syncthreads();

  int tile_n = blockIdx.y * 4 + wave;          // 8 n-tiles (115 -> 128 padded)
  int half = lane >> 4, rc = lane & 15;
  int n = tile_n * 16 + rc;
  v8f acc = {};
  for (int kb = 0; kb < C_DIM; kb += 32) {
    v16h af, bf;
#pragma unroll
    for (int e = 0; e < 16; ++e)
      af[e] = (_Float16)sA[rc * C_DIM + kb + ak_off(e, half)];
#pragma unroll
    for (int e = 0; e < 16; ++e)
      bf[e] = (_Float16)((n < HID_DIM) ? w[n * C_DIM + kb + bk_off(e, half)] : 0.0f);
    acc = wmma_f16(af, bf, acc);
  }
  if (n < HID_DIM) {
    float inv = g[n] * rsqrtf(vv[n] + 1e-5f);
    float sh = be[n] - mm[n] * inv;
#pragma unroll
    for (int r = 0; r < 8; ++r) {
      int m = m0 + r + 8 * half;
      float v = acc[r] * inv + sh;
      h1[m * HID_DIM + n] = v * sigf(v);
    }
  }
}

// ============ K8: mlp2 GEMM [16384 x 115] x [115 x 96] + BN2 + residual ====
// H1 rows (stride 115 floats) aren't 16B aligned -> async B32 staging.
__global__ void k_mlp2(const float* __restrict__ h1,
                       const float* __restrict__ w,
                       const float* __restrict__ g,
                       const float* __restrict__ be,
                       const float* __restrict__ mm,
                       const float* __restrict__ vv,
                       const float* __restrict__ x2,
                       float* __restrict__ out) {
  __shared__ float sA[HID_DIM * 16];           // [k][row], 7.2 KB
  int lane = threadIdx.x & 31, wave = threadIdx.x >> 5;
  int tile_m = blockIdx.x;
  int m0 = tile_m * 16;
  int b = m0 >> 12;
  int lbase = m0 & (L_TOT - 1);

  for (int i = threadIdx.x; i < HID_DIM * 16; i += 128) {
    int d = i >> 4, row = i & 15;
    async_ld_b32(&sA[i], &h1[(m0 + row) * HID_DIM + d]);
  }
  wait_async0();
  __syncthreads();

  int tile_n = blockIdx.y * 4 + wave;
  if (tile_n >= 6) return;
  int half = lane >> 4, rc = lane & 15;
  int n = tile_n * 16 + rc;
  v8f acc = {};
  for (int kb = 0; kb < 128; kb += 32) {       // K = 115 padded to 128
    v16h af, bf;
#pragma unroll
    for (int e = 0; e < 16; ++e) {
      int k = kb + ak_off(e, half);
      af[e] = (_Float16)((k < HID_DIM) ? sA[k * 16 + rc] : 0.0f);
    }
#pragma unroll
    for (int e = 0; e < 16; ++e) {
      int k = kb + bk_off(e, half);
      bf[e] = (_Float16)((k < HID_DIM) ? w[n * HID_DIM + k] : 0.0f);
    }
    acc = wmma_f16(af, bf, acc);
  }
  float inv = g[n] * rsqrtf(vv[n] + 1e-5f);
  float sh = be[n] - mm[n] * inv;
#pragma unroll
  for (int r = 0; r < 8; ++r) {
    int m = m0 + r + 8 * half;
    int l = lbase + r + 8 * half;
    out[(b * C_DIM + n) * L_TOT + l] = x2[m * C_DIM + n] + acc[r] * inv + sh;
  }
}

// ---------------- host-side launch ----------------
extern "C" void kernel_launch(void* const* d_in, const int* in_sizes, int n_in,
                              void* d_out, int out_size, void* d_ws, size_t ws_size,
                              hipStream_t stream) {
  (void)in_sizes; (void)n_in; (void)out_size; (void)ws_size;
  const float* x        = (const float*)d_in[0];
  const float* inw      = (const float*)d_in[1];
  const float* inb      = (const float*)d_in[2];
  const float* convw    = (const float*)d_in[3];
  const float* convb    = (const float*)d_in[4];
  const float* xprojw   = (const float*)d_in[5];
  const float* dtw      = (const float*)d_in[6];
  const float* dtb      = (const float*)d_in[7];
  const float* alog     = (const float*)d_in[8];
  const float* dsv      = (const float*)d_in[9];
  const float* lng      = (const float*)d_in[10];
  const float* lnb      = (const float*)d_in[11];
  const float* outw     = (const float*)d_in[12];
  const float* outb     = (const float*)d_in[13];
  const float* mlpw1    = (const float*)d_in[14];
  const float* bn1g     = (const float*)d_in[15];
  const float* bn1b     = (const float*)d_in[16];
  const float* bn1m     = (const float*)d_in[17];
  const float* bn1v     = (const float*)d_in[18];
  const float* mlpw2    = (const float*)d_in[19];
  const float* bn2g     = (const float*)d_in[20];
  const float* bn2b     = (const float*)d_in[21];
  const float* bn2m     = (const float*)d_in[22];
  const float* bn2v     = (const float*)d_in[23];
  float* out = (float*)d_out;

  // workspace layout (floats)
  float* ws    = (float*)d_ws;
  float* X1pre = ws;                          // B*D*L   = 3145728
  float* X1    = X1pre + 3145728;             // 3145728
  float* Z     = X1    + 3145728;             // B*L*D   = 3145728
  float* XDBL  = Z     + 3145728;             // B*K*38*L= 2490368
  float* YS    = XDBL  + 2490368;             // B*K*D*L = 12582912
  float* Y     = YS    + 12582912;            // B*L*D   = 3145728
  float* X2    = Y     + 3145728;             // B*L*C   = 1572864
  float* H1    = X1pre;                       // reuse: B*L*HID = 1884160 <= 3145728

  const int MT = (B_DIM * L_TOT) / 16;        // 1024 M-tiles

  // 1) in_proj GEMM -> x1pre (NCHW), z (B,L,D)
  k_inproj<<<dim3(MT, 6), 128, 0, stream>>>(x, inw, inb, X1pre, Z);
  // 2) depthwise conv + SiLU -> X1 (B,D,L)
  k_conv<<<(B_DIM * D_DIM * L_TOT + 255) / 256, 256, 0, stream>>>(X1pre, convw, convb, X1);
  // 3) x_proj GEMM per direction -> XDBL (B,K,38,L)
  k_xproj<<<dim3(MT, 1, K_DIR), 128, 0, stream>>>(X1, xprojw, XDBL);
  // 4) selective scan (dt fused) -> YS (B,K,D,L)
  k_scan<<<(B_DIM * K_DIR * D_DIM + 127) / 128, 128, 0, stream>>>(
      X1, XDBL, dtw, dtb, alog, dsv, YS);
  // 5) combine + LN + gate -> Y (B,L,D)
  k_combine<<<(B_DIM * L_TOT) / 8, 256, 0, stream>>>(YS, Z, lng, lnb, Y);
  // 6) out_proj GEMM + residual -> X2 (B,L,C)
  k_outproj<<<dim3(MT, 2), 128, 0, stream>>>(Y, outw, outb, x, X2);
  // 7) mlp1 GEMM + BN1 + SiLU -> H1 (B,L,HID)  (reuses X1pre region)
  k_mlp1<<<dim3(MT, 2), 128, 0, stream>>>(X2, mlpw1, bn1g, bn1b, bn1m, bn1v, H1);
  // 8) mlp2 GEMM + BN2 + residual -> out (B,C,H,W)
  k_mlp2<<<dim3(MT, 2), 128, 0, stream>>>(H1, mlpw2, bn2g, bn2b, bn2m, bn2v, X2, out);
}
